// LayerNormLSTM_60756607369774
// MI455X (gfx1250) — matrix operand order repaired and checked by reading, not compiled, not run on hardware
//
#include <hip/hip_runtime.h>
#include <math.h>

// ---------------------------------------------------------------------------
// LayerNorm-LSTM for MI455X (gfx1250): bf16 WMMA GEMMs + TDM-staged A tiles
// + fused per-step cell. T=512, B=64, I=H=512, L=2, G=4H=2048.
// ---------------------------------------------------------------------------

#define T_LEN 512
#define BB    64
#define II    512
#define HH    512
#define LL    2
#define GG    2048
#define EPS_LN 1e-5f
#define LDSP  520          // LDS row pitch in bf16 elems: 512 + 8 (16B bank skew)

typedef __attribute__((ext_vector_type(16))) __bf16 v16bf;
typedef __attribute__((ext_vector_type(8)))  float  v8f;
typedef __attribute__((ext_vector_type(4)))  unsigned int v4u;
typedef __attribute__((ext_vector_type(8)))  int v8i;
typedef __attribute__((ext_vector_type(4)))  int v4i;

#if defined(__has_builtin)
#if __has_builtin(__builtin_amdgcn_tensor_load_to_lds)
#define HAVE_TDM 1
#endif
#endif

struct __align__(16) B128 { unsigned int w[4]; };
union FragU { v16bf v; B128 q[2]; };

// A-matrix 16x32 bf16 fragment (ISA 7.12.2): lane holds row m = lane&15;
// K = {klo..klo+7, 16+klo..16+klo+7}, klo = (lane&16)?8:0.
__device__ __forceinline__ v16bf load_a_frag(const unsigned short* p, int lane) {
    FragU f;
    const int klo = (lane & 16) ? 8 : 0;
    f.q[0] = *reinterpret_cast<const B128*>(p + klo);
    f.q[1] = *reinterpret_cast<const B128*>(p + 16 + klo);
    return f.v;
}

// B-matrix 32x16 bf16 fragment: lane holds col n = lane&15;
// K = {koff..koff+7, 16+koff..16+koff+7}, koff = (lane&16)?8:0.
__device__ __forceinline__ v16bf load_b_frag(const unsigned short* p, int lane) {
    FragU f;
    const int koff = (lane & 16) ? 8 : 0;
    f.q[0] = *reinterpret_cast<const B128*>(p + koff);
    f.q[1] = *reinterpret_cast<const B128*>(p + 16 + koff);
    return f.v;
}

__device__ __forceinline__ unsigned short f32_to_bf16(float f) {
    unsigned int u = __float_as_uint(f);
    u += 0x7fffu + ((u >> 16) & 1u);          // round-to-nearest-even
    return (unsigned short)(u >> 16);
}

__device__ __forceinline__ float sigmoid_f(float x) { return 1.f / (1.f + expf(-x)); }

// ---------------------------------------------------------------------------
// TDM: DMA a 16-row x 512-elem bf16 tile (row stride 512) from global into
// LDS with 16B padding per 1024B row (pad_interval=256 DWORDs, pad_amount=4
// DWORDs) -> LDS pitch 1040B = LDSP bf16 elems.  D# per CDNA5 ISA ch.8.
// This toolchain exposes the 6-arg builtin form:
//   (uint32x4 g0, int32x8 g1, int32x4 g2, int32x4 g3, int32x8 xtra, i32 cpol)
// ---------------------------------------------------------------------------
#ifdef HAVE_TDM
__device__ __forceinline__ void tdm_load_16x512(const unsigned short* gsrc,
                                                unsigned lds_off) {
    const unsigned long long ga = (unsigned long long)(uintptr_t)gsrc;
    v4u g0;
    g0[0] = 1u;                                             // count=1, user mode
    g0[1] = lds_off;                                        // lds_addr (bytes)
    g0[2] = (unsigned)(ga & 0xffffffffu);                   // global_addr[31:0]
    g0[3] = (unsigned)((ga >> 32) & 0x1ffffffu)             // global_addr[56:32]
          | (2u << 30);                                     // type = 2 (image)
    v8i g1;
    g1[0] = (int)((1u << 16)                                // data_size: 2 bytes
          | (1u << 20)                                      // pad_enable
          | (7u << 22)                                      // pad_interval: 256 DW
          | (3u << 25));                                    // pad_amount: 4 DW
    g1[1] = (int)((512u & 0xffffu) << 16);                  // tensor_dim0[15:0]
    g1[2] = (int)((512u >> 16) | (16u << 16));              // dim0 hi | tensor_dim1 lo
    g1[3] = (int)((16u >> 16) | (512u << 16));              // dim1 hi | tile_dim0
    g1[4] = 16;                                             // tile_dim1=16, tile_dim2=0
    g1[5] = 512;                                            // tensor_dim0_stride lo32
    g1[6] = 0;
    g1[7] = 0;
    const v4i z4 = {0, 0, 0, 0};
    const v8i z8 = {0, 0, 0, 0, 0, 0, 0, 0};
    __builtin_amdgcn_tensor_load_to_lds(g0, g1, z4, z4, z8, 0);
}
#endif

// Stage 16x512 bf16 tile into padded LDS (pitch LDSP). Caller must
// __syncthreads() afterwards.
__device__ __forceinline__ void stage_tile_16x512(const unsigned short* gsrc,
                                                  unsigned short* lds, int tid) {
#ifdef HAVE_TDM
    if (tid == 0) {
        tdm_load_16x512(gsrc, (unsigned)(uintptr_t)lds);
        __builtin_amdgcn_s_wait_tensorcnt(0);
    }
#else
    for (int idx = tid; idx < 1024; idx += 256) {           // 16B chunks
        const int r = idx >> 6, ck = idx & 63;
        *reinterpret_cast<B128*>(&lds[r * LDSP + ck * 8]) =
            *reinterpret_cast<const B128*>(&gsrc[r * 512 + ck * 8]);
    }
#endif
}

// ---------------------------------------------------------------------------
// f32 -> bf16 conversion / zero fill
// ---------------------------------------------------------------------------
__global__ __launch_bounds__(256) void cvt_bf16_kernel(const float* __restrict__ src,
                                                       unsigned short* __restrict__ dst,
                                                       int n) {
    int i = blockIdx.x * 256 + threadIdx.x;
    if (i < n) dst[i] = f32_to_bf16(src[i]);
}

__global__ __launch_bounds__(256) void zero_f32_kernel(float* __restrict__ p, int n) {
    int i = blockIdx.x * 256 + threadIdx.x;
    if (i < n) p[i] = 0.f;
}

// ---------------------------------------------------------------------------
// gx GEMM: out[r,g] = sum_k A[r,k]*W[g,k] + bias[g]   (raw, LN applied after)
// A: [R,K] bf16.  W: [G,K] bf16.  out: [R,G] f32.
// Block = 8 waves sharing one 16-row A tile (TDM-staged in LDS); each wave
// owns a 16x64 column slice (4 WMMA tiles). B fragments are software-
// pipelined one tile ahead so waits don't drain to zero before each WMMA.
// ---------------------------------------------------------------------------
__global__ __launch_bounds__(256) void gemm_gx_kernel(
    const unsigned short* __restrict__ A,
    const unsigned short* __restrict__ W,
    const float* __restrict__ bias,
    float* __restrict__ out,
    int K, int G)
{
    __shared__ unsigned short ldsA[16 * LDSP];
    const int lane = threadIdx.x & 31;
    const int wave = threadIdx.x >> 5;
    const int nlo  = lane & 15;
    const int row0 = (blockIdx.x >> 2) * 16;               // 4 blocks per row tile
    const int col0 = ((blockIdx.x & 3) * 8 + wave) * 64;

    stage_tile_16x512(A + (size_t)row0 * K, ldsA, threadIdx.x);
    __syncthreads();

    const unsigned short* bb[4];
#pragma unroll
    for (int j = 0; j < 4; ++j)
        bb[j] = W + (size_t)(col0 + j * 16 + nlo) * K;

    v8f acc[4] = {};
    v16bf bcur = load_b_frag(bb[0], lane);
    for (int k0 = 0; k0 < K; k0 += 32) {
        v16bf a = load_a_frag(&ldsA[nlo * LDSP + k0], lane);
#pragma unroll
        for (int j = 0; j < 4; ++j) {
            v16bf bnx = bcur;
            if (j < 3)               bnx = load_b_frag(bb[j + 1] + k0, lane);
            else if (k0 + 32 < K)    bnx = load_b_frag(bb[0] + k0 + 32, lane);
            acc[j] = __builtin_amdgcn_wmma_f32_16x16x32_bf16(
                false, a, false, bcur, (short)0, acc[j], false, false);
            bcur = bnx;
        }
    }

    const int mhalf = (lane & 16) ? 8 : 0;
#pragma unroll
    for (int j = 0; j < 4; ++j) {
        const int col = col0 + j * 16 + nlo;
        const float bv = bias[col];
#pragma unroll
        for (int v = 0; v < 8; ++v) {
            const int row = row0 + mhalf + v;
            out[(size_t)row * G + col] = acc[j][v] + bv;
        }
    }
}

// ---------------------------------------------------------------------------
// Row LayerNorm in place: p[row, 0..N) -> (x-mean)*rstd*gam + bet
// ---------------------------------------------------------------------------
__global__ __launch_bounds__(256) void ln_rows_kernel(
    float* __restrict__ data, int N,
    const float* __restrict__ gam, const float* __restrict__ bet)
{
    float* p = data + (size_t)blockIdx.x * N;
    const int tid = threadIdx.x;
    float s = 0.f, s2 = 0.f;
    for (int i = tid; i < N; i += 256) { float v = p[i]; s += v; s2 += v * v; }
    __shared__ float ss[256], ss2[256];
    ss[tid] = s; ss2[tid] = s2;
    __syncthreads();
    for (int st = 128; st > 0; st >>= 1) {
        if (tid < st) { ss[tid] += ss[tid + st]; ss2[tid] += ss2[tid + st]; }
        __syncthreads();
    }
    const float mean = ss[0] / (float)N;
    const float var  = ss2[0] / (float)N - mean * mean;
    const float rstd = rsqrtf(var + EPS_LN);
    for (int i = tid; i < N; i += 256)
        p[i] = (p[i] - mean) * rstd * gam[i] + bet[i];
}

// ---------------------------------------------------------------------------
// Fused LSTM step (one launch per (layer, t)):
//   raw   = h_{t-1} @ Whh^T + bhh            (bf16 WMMA, f32 acc)
//   gates = gx_t + LN_G(raw) * ghh + bhh_ln
//   i,f,o = sigmoid, g = tanh; c = f*c + i*g; h = o * tanh(LN_H(c)*gho + bho)
// Grid: B/16 = 4 blocks x 256 threads. h_{t-1} tile TDM-staged in LDS.
// Each wave owns hidden units [w*64, w*64+64) and the 4 matching gate column
// blocks (i/f/o/g) -> cell update is wave-local; the two LayerNorm
// reductions cross waves via ds_add_f32 + barriers.
// ---------------------------------------------------------------------------
__global__ __launch_bounds__(256) void lstm_step_kernel(
    const unsigned short* __restrict__ hprev,   // [B,H] bf16, nullptr if t==0
    const unsigned short* __restrict__ Whh,     // [G,H] bf16
    const float* __restrict__ bhh,              // [G]
    const float* __restrict__ ghh,  const float* __restrict__ bhh_ln,   // [G]
    const float* __restrict__ gho,  const float* __restrict__ bho_ln,   // [H]
    const float* __restrict__ gx_t,             // [B,G] f32 (LN'd, + b_ih)
    float* __restrict__ cbuf,                   // [B,H] f32 (state)
    unsigned short* __restrict__ hout,          // [B,H] bf16 (seq slot t)
    float* __restrict__ y_out,                  // [B,H] f32 or nullptr
    float* __restrict__ hy_out,                 // [B,H] f32 or nullptr (t==T-1)
    float* __restrict__ cy_out)                 // [B,H] f32 or nullptr (t==T-1)
{
    const int lane = threadIdx.x & 31;
    const int w    = threadIdx.x >> 5;          // wave 0..7
    const int row0 = blockIdx.x * 16;           // batch rows [row0, row0+16)
    const int mhalf = (lane & 16) ? 8 : 0;
    const int nlo = lane & 15;

    __shared__ unsigned short ldsA[16 * LDSP];
    __shared__ float rs[16], rs2[16];           // gate-LN row sums (over G)
    __shared__ float cs[16], cs2[16];           // cell-LN row sums (over H)
    if (threadIdx.x < 16) { rs[threadIdx.x] = 0.f; rs2[threadIdx.x] = 0.f;
                            cs[threadIdx.x] = 0.f; cs2[threadIdx.x] = 0.f; }

    // ---- GEMM: acc[ga*4+j] covers cols ga*512 + w*64 + j*16 + nlo ----------
    v8f acc[16] = {};
    if (hprev != nullptr) {
        stage_tile_16x512(hprev + (size_t)row0 * HH, ldsA, threadIdx.x);
        __syncthreads();

        const unsigned short* bb[16];
#pragma unroll
        for (int tk = 0; tk < 16; ++tk) {
            const int col = (tk >> 2) * HH + w * 64 + (tk & 3) * 16 + nlo;
            bb[tk] = Whh + (size_t)col * HH;
        }
        v16bf bcur = load_b_frag(bb[0], lane);
        for (int k0 = 0; k0 < HH; k0 += 32) {
            v16bf a = load_a_frag(&ldsA[nlo * LDSP + k0], lane);
#pragma unroll
            for (int tk = 0; tk < 16; ++tk) {
                v16bf bnx = bcur;
                if (tk < 15)              bnx = load_b_frag(bb[tk + 1] + k0, lane);
                else if (k0 + 32 < HH)    bnx = load_b_frag(bb[0] + k0 + 32, lane);
                acc[tk] = __builtin_amdgcn_wmma_f32_16x16x32_bf16(
                    false, a, false, bcur, (short)0, acc[tk], false, false);
                bcur = bnx;
            }
        }
    }
    __syncthreads();   // also covers rs/cs init visibility

    // ---- add b_hh, accumulate per-row sum / sumsq over G -------------------
    float ps[8] = {}, ps2[8] = {};
#pragma unroll
    for (int ga = 0; ga < 4; ++ga) {
#pragma unroll
        for (int j = 0; j < 4; ++j) {
            const int col = ga * HH + w * 64 + j * 16 + nlo;
            const float bv = bhh[col];
#pragma unroll
            for (int v = 0; v < 8; ++v) {
                float x = acc[ga * 4 + j][v] + bv;
                acc[ga * 4 + j][v] = x;
                ps[v] += x; ps2[v] += x * x;
            }
        }
    }
#pragma unroll
    for (int v = 0; v < 8; ++v) {
        atomicAdd(&rs[mhalf + v], ps[v]);
        atomicAdd(&rs2[mhalf + v], ps2[v]);
    }
    __syncthreads();

    float gmean[8], grstd[8];
#pragma unroll
    for (int v = 0; v < 8; ++v) {
        const int m = mhalf + v;
        const float mu = rs[m] * (1.f / (float)GG);
        const float va = rs2[m] * (1.f / (float)GG) - mu * mu;
        gmean[v] = mu; grstd[v] = rsqrtf(va + EPS_LN);
    }

    // ---- LN + gx + activations (in place) ----------------------------------
#pragma unroll
    for (int ga = 0; ga < 4; ++ga) {
#pragma unroll
        for (int j = 0; j < 4; ++j) {
            const int col = ga * HH + w * 64 + j * 16 + nlo;
            const float gam = ghh[col], bet = bhh_ln[col];
#pragma unroll
            for (int v = 0; v < 8; ++v) {
                const int brow = row0 + mhalf + v;
                float n = (acc[ga * 4 + j][v] - gmean[v]) * grstd[v] * gam + bet;
                float gate = gx_t[(size_t)brow * GG + col] + n;
                acc[ga * 4 + j][v] = (ga == 3) ? tanhf(gate) : sigmoid_f(gate);
            }
        }
    }

    // ---- cell update: c = f*c + i*g;  accumulate cell-LN sums over H -------
    float pc[8] = {}, pc2[8] = {};
#pragma unroll
    for (int j = 0; j < 4; ++j) {
        const int hu = w * 64 + j * 16 + nlo;
#pragma unroll
        for (int v = 0; v < 8; ++v) {
            const int brow = row0 + mhalf + v;
            const float cold = cbuf[(size_t)brow * HH + hu];
            const float cn = acc[4 + j][v] * cold + acc[0 + j][v] * acc[12 + j][v];
            acc[0 + j][v] = cn;                // slot 0 now holds new c
            pc[v] += cn; pc2[v] += cn * cn;
        }
    }
#pragma unroll
    for (int v = 0; v < 8; ++v) {
        atomicAdd(&cs[mhalf + v], pc[v]);
        atomicAdd(&cs2[mhalf + v], pc2[v]);
    }
    __syncthreads();

    float cmean[8], crstd[8];
#pragma unroll
    for (int v = 0; v < 8; ++v) {
        const int m = mhalf + v;
        const float mu = cs[m] * (1.f / (float)HH);
        const float va = cs2[m] * (1.f / (float)HH) - mu * mu;
        cmean[v] = mu; crstd[v] = rsqrtf(va + EPS_LN);
    }

    // ---- h = o * tanh(LN(c));  write state + outputs -----------------------
#pragma unroll
    for (int j = 0; j < 4; ++j) {
        const int hu = w * 64 + j * 16 + nlo;
        const float gam = gho[hu], bet = bho_ln[hu];
#pragma unroll
        for (int v = 0; v < 8; ++v) {
            const int brow = row0 + mhalf + v;
            const size_t idx = (size_t)brow * HH + hu;
            const float cn = acc[0 + j][v];
            const float h = acc[8 + j][v] * tanhf((cn - cmean[v]) * crstd[v] * gam + bet);
            cbuf[idx] = cn;
            hout[idx] = f32_to_bf16(h);
            if (y_out)  y_out[idx] = h;
            if (hy_out) { hy_out[idx] = h; cy_out[idx] = cn; }
        }
    }
}

// ---------------------------------------------------------------------------
// Host driver
// ---------------------------------------------------------------------------
extern "C" void kernel_launch(void* const* d_in, const int* in_sizes, int n_in,
                              void* d_out, int out_size, void* d_ws, size_t ws_size,
                              hipStream_t stream) {
    (void)in_sizes; (void)n_in; (void)out_size; (void)ws_size;
    const float* x       = (const float*)d_in[0];   // [T,B,I]
    const float* W_ih    = (const float*)d_in[1];   // [L,G,I]
    const float* W_hh    = (const float*)d_in[2];   // [L,G,H]
    const float* b_ih    = (const float*)d_in[3];   // [L,G]
    const float* b_hh    = (const float*)d_in[4];   // [L,G]
    const float* g_ih    = (const float*)d_in[5];   // [L,G]
    const float* beta_ih = (const float*)d_in[6];   // [L,G]
    const float* g_hh    = (const float*)d_in[7];   // [L,G]
    const float* beta_hh = (const float*)d_in[8];   // [L,G]
    const float* g_ho    = (const float*)d_in[9];   // [L,H]
    const float* beta_ho = (const float*)d_in[10];  // [L,H]

    // workspace carve-up
    char* ws = (char*)d_ws;
    unsigned short* seq_bf = (unsigned short*)ws;                 // T*B*H bf16
    size_t off = (size_t)T_LEN * BB * HH * 2;
    unsigned short* Wih_bf = (unsigned short*)(ws + off);         // L*G*I bf16
    off += (size_t)LL * GG * II * 2;
    unsigned short* Whh_bf = (unsigned short*)(ws + off);         // L*G*H bf16
    off += (size_t)LL * GG * HH * 2;
    float* gx = (float*)(ws + off);                               // T*B*G f32
    off += (size_t)T_LEN * BB * GG * 4;
    float* cbuf = (float*)(ws + off);                             // B*H f32

    float* y_base  = (float*)d_out;                               // [T,B,H]
    float* hy_base = y_base + (size_t)T_LEN * BB * HH;            // [L,B,H]
    float* cy_base = hy_base + (size_t)LL * BB * HH;              // [L,B,H]

    // f32 -> bf16 conversions
    {
        int n = T_LEN * BB * II;
        cvt_bf16_kernel<<<(n + 255) / 256, 256, 0, stream>>>(x, seq_bf, n);
        n = LL * GG * II;
        cvt_bf16_kernel<<<(n + 255) / 256, 256, 0, stream>>>(W_ih, Wih_bf, n);
        n = LL * GG * HH;
        cvt_bf16_kernel<<<(n + 255) / 256, 256, 0, stream>>>(W_hh, Whh_bf, n);
    }

    const int R = T_LEN * BB;                     // 32768 sequence rows
    for (int l = 0; l < LL; ++l) {
        // time-parallel input-path GEMM + bias, then row LayerNorm
        const int nBlocks = (R / 16) * (GG / 64) / 8;   // 4 blocks per row tile
        gemm_gx_kernel<<<nBlocks, 256, 0, stream>>>(
            seq_bf, Wih_bf + (size_t)l * GG * II, b_ih + (size_t)l * GG,
            gx, II, GG);
        ln_rows_kernel<<<R, 256, 0, stream>>>(
            gx, GG, g_ih + (size_t)l * GG, beta_ih + (size_t)l * GG);

        // c0 = 0
        {
            int n = BB * HH;
            zero_f32_kernel<<<(n + 255) / 256, 256, 0, stream>>>(cbuf, n);
        }

        // sequential scan; h_t overwrites seq slot t (gx already consumed input)
        for (int t = 0; t < T_LEN; ++t) {
            lstm_step_kernel<<<BB / 16, 256, 0, stream>>>(
                (t > 0) ? seq_bf + (size_t)(t - 1) * BB * HH : nullptr,
                Whh_bf + (size_t)l * GG * HH,
                b_hh + (size_t)l * GG,
                g_hh + (size_t)l * GG, beta_hh + (size_t)l * GG,
                g_ho + (size_t)l * HH, beta_ho + (size_t)l * HH,
                gx + (size_t)t * BB * GG,
                cbuf,
                seq_bf + (size_t)t * BB * HH,
                (l == LL - 1) ? y_base + (size_t)t * BB * HH : nullptr,
                (t == T_LEN - 1) ? hy_base + (size_t)l * BB * HH : nullptr,
                (t == T_LEN - 1) ? cy_base + (size_t)l * BB * HH : nullptr);
        }
    }
}
